// ArcFaceFC_17910013624961
// MI455X (gfx1250) — compile-verified, matrix operand order checked by reading, not run on hardware
//
#include <hip/hip_runtime.h>
#include <hip/hip_bf16.h>
#include <math.h>

#define N_ROWS 1024
#define DIM    512
#define NCLS   100000
#define BC     128                      // columns (classes) per workgroup
#define NWG    ((NCLS + BC - 1) / BC)   // 782
#define STRIDE 520                      // padded bf16 row stride in LDS (512 + 8)
#define RBLK   128                      // rows per outer iteration
#define ARC_SCALE  64.0f
#define ARC_MARGIN 0.5f
#define ARC_EPS    1e-12f

typedef __attribute__((ext_vector_type(16))) __bf16 v16bf;
typedef __attribute__((ext_vector_type(8)))  float  v8f;
typedef __attribute__((ext_vector_type(4)))  unsigned int v4u;
typedef __attribute__((ext_vector_type(8)))  int  v8i;
typedef __attribute__((ext_vector_type(4)))  int  v4i;

// LDS bytes: B chunk [BC][STRIDE] + A block [RBLK][STRIDE] + stats 2x[2][RBLK]
#define SMEM_BYTES ((size_t)(BC + RBLK) * STRIDE * 2 + 2 * RBLK * 4 * 2)

__device__ __forceinline__ float waveReduceSum32(float v) {
#pragma unroll
  for (int m = 16; m >= 1; m >>= 1) v += __shfl_xor(v, m, 32);
  return v;
}

// ---------------------------------------------------------------------------
// TDM: load a [RBLK x DIM] bf16 tile (row-major, tensor = [N_ROWS x DIM])
// into LDS at byte offset ldsOff, with HW padding of 4 dwords every 256 dwords
// so the LDS layout has the 520-element padded stride for free.
// ---------------------------------------------------------------------------
#if __has_builtin(__builtin_amdgcn_tensor_load_to_lds)
#define HAVE_TDM 1
__device__ __forceinline__ void tdm_load_A(const __bf16* gsrc,
                                           unsigned ldsOff) {
  const unsigned long long ga = (unsigned long long)(size_t)gsrc;
  v4u g0 = {};
  g0.x = 1u;                                        // count=1, user mode
  g0.y = ldsOff;                                    // lds_addr (bytes)
  g0.z = (unsigned)(ga & 0xFFFFFFFFu);              // global_addr[31:0]
  g0.w = (unsigned)((ga >> 32) & 0x01FFFFFFu)       // global_addr[56:32]
         | 0x80000000u;                             // type=2 ("image")
  v8i g1 = {};
  // wgmask=0 | data_size=1(2B)<<16 | pad_enable<<20 | pad_interval=7(<<22)
  // | pad_amount=3(<<25)  -> pad 4 dwords after every 256 dwords (one row)
  g1[0] = (int)((1u << 16) | (1u << 20) | (7u << 22) | (3u << 25));
  g1[1] = (int)((unsigned)(DIM & 0xFFFF) << 16);    // tensor_dim0[15:0]
  g1[2] = (int)((unsigned)(N_ROWS & 0xFFFF) << 16); // dim0 hi=0 | dim1[15:0]
  g1[3] = (int)((unsigned)DIM << 16);               // dim1 hi=0 | tile_dim0
  g1[4] = (int)RBLK;                                // tile_dim1 | tile_dim2=0
  g1[5] = (int)DIM;                                 // tensor_dim0_stride lo32
  g1[6] = 0;                                        // stride0 hi | stride1 lo
  g1[7] = 0;
  v4i g2 = {};
  v4i g3 = {};
#if __clang_major__ >= 23
  v8i g4 = {};
  __builtin_amdgcn_tensor_load_to_lds(g0, g1, g2, g3, g4, 0);
#else
  __builtin_amdgcn_tensor_load_to_lds(g0, g1, g2, g3, 0);
#endif
}
#else
#define HAVE_TDM 0
#endif

// ---------------------------------------------------------------------------
// Kernel 1: L2-normalize images, emit bf16 A matrix [N_ROWS x DIM]
// ---------------------------------------------------------------------------
__global__ void k_norm_images(const float* __restrict__ img,
                              __bf16* __restrict__ neb) {
  const int wave = threadIdx.x >> 5, lane = threadIdx.x & 31;
  const int n = blockIdx.x * 8 + wave;
  const float* row = img + (size_t)n * DIM;
  float vals[16];
  float ss = 0.f;
#pragma unroll
  for (int i = 0; i < 16; ++i) {
    float x = row[lane + 32 * i];
    vals[i] = x;
    ss += x * x;
  }
  ss = waveReduceSum32(ss);
  const float rn = 1.0f / fmaxf(sqrtf(ss), ARC_EPS);
  __bf16* orow = neb + (size_t)n * DIM;
#pragma unroll
  for (int i = 0; i < 16; ++i) orow[lane + 32 * i] = (__bf16)(vals[i] * rn);
}

// ---------------------------------------------------------------------------
// Kernel 1b: exact f32 target logits per row. Emits BOTH the plain scaled
// target cosine (tplain) and the margin-applied one (tmarg); the softmax
// swap (replace tplain by tmarg inside the denominator) is done in k_combine
// so the hot GEMM epilogue needs no labels, compares, or trig.
// ---------------------------------------------------------------------------
__global__ void k_target_logit(const float* __restrict__ img,
                               const int* __restrict__ labels,
                               const float* __restrict__ weight,
                               float* __restrict__ tplain,
                               float* __restrict__ tmarg) {
  const int wave = threadIdx.x >> 5, lane = threadIdx.x & 31;
  const int n = blockIdx.x * 8 + wave;
  const int lab = labels[n];
  const int labc = lab < 0 ? 0 : lab;
  const float* ir = img + (size_t)n * DIM;
  const float* wr = weight + (size_t)labc * DIM;
  float di = 0.f, si = 0.f, sw = 0.f;
#pragma unroll
  for (int i = 0; i < 16; ++i) {
    float a = ir[lane + 32 * i];
    float b = wr[lane + 32 * i];
    di += a * b;
    si += a * a;
    sw += b * b;
  }
  di = waveReduceSum32(di);
  si = waveReduceSum32(si);
  sw = waveReduceSum32(sw);
  if (lane == 0) {
    float t = di / (fmaxf(sqrtf(si), ARC_EPS) * fmaxf(sqrtf(sw), ARC_EPS));
    t = fminf(fmaxf(t, -1.0f), 1.0f);
    const float tp = t * ARC_SCALE;
    const float tm =
        (lab != -1) ? cosf(acosf(t) + ARC_MARGIN) * ARC_SCALE : tp;
    tplain[n] = tp;
    tmarg[n] = tm;
  }
}

// ---------------------------------------------------------------------------
// Kernel 2 (main): fused weight-normalize + bf16 WMMA GEMM (2x4 register
// tiling: 1.5 LDS b128 loads per WMMA) + plain online-softmax partials.
// WG = 256 threads = 8 waves = 4 row-groups x 2 col-groups; owns BC=128 cols.
// A block staged by the Tensor Data Mover (HW-padded LDS stride).
// ---------------------------------------------------------------------------
__global__ void k_arcface_main(const float* __restrict__ weight,
                               const __bf16* __restrict__ neb,
                               float* __restrict__ pmax,
                               float* __restrict__ psum) {
  extern __shared__ char smem[];
  __bf16* bLds = (__bf16*)smem;                 // [BC][STRIDE]
  __bf16* aLds = bLds + (size_t)BC * STRIDE;    // [RBLK][STRIDE]
  float* wMax = (float*)(aLds + RBLK * STRIDE); // [2][RBLK]
  float* wSum = wMax + 2 * RBLK;                // [2][RBLK]

  const int tid = threadIdx.x;
  const int wave = tid >> 5, lane = tid & 31;
  const int c0 = blockIdx.x * BC;

  // ---- Phase 1: normalize this WG's weight chunk into LDS as bf16 ----
  for (int j = wave; j < BC; j += 8) {
    const int c = c0 + j;
    float vals[16];
    float ss = 0.f;
    if (c < NCLS) {
      const float* wr = weight + (size_t)c * DIM;
#pragma unroll
      for (int i = 0; i < 16; ++i) {
        float x = wr[lane + 32 * i];
        vals[i] = x;
        ss += x * x;
      }
    } else {
#pragma unroll
      for (int i = 0; i < 16; ++i) vals[i] = 0.f;
    }
    ss = waveReduceSum32(ss);
    const float rn = (c < NCLS) ? (1.0f / fmaxf(sqrtf(ss), ARC_EPS)) : 0.f;
#pragma unroll
    for (int i = 0; i < 16; ++i)
      bLds[j * STRIDE + lane + 32 * i] = (__bf16)(vals[i] * rn);
  }
  __syncthreads();

  const int nl = lane & 15;
  const int half = lane >> 4;
  const int rwave = wave >> 1;  // 0..3 row group (32 rows each)
  const int cwave = wave & 1;   // 0..1 col group (64 cols each)
  const __bf16* aP[2];
#pragma unroll
  for (int rt = 0; rt < 2; ++rt)
    aP[rt] = aLds + (rwave * 32 + rt * 16 + nl) * STRIDE + half * 16;
  const __bf16* bP[4];
  int col[4];
#pragma unroll
  for (int ct = 0; ct < 4; ++ct) {
    bP[ct] = bLds + (cwave * 64 + ct * 16 + nl) * STRIDE + half * 16;
    col[ct] = c0 + cwave * 64 + ct * 16 + nl;
  }
  const unsigned aOff = (unsigned)((size_t)BC * STRIDE * 2);  // aLds byte off

  // ---- Phase 2: 8 outer iterations of 128 rows each ----
  for (int ob = 0; ob < N_ROWS / RBLK; ++ob) {
#if HAVE_TDM
    if (wave == 0) {  // one TDM descriptor stages the whole 128x512 A block
      tdm_load_A(neb + (size_t)(ob * RBLK) * DIM,
                 (unsigned)(size_t)smem + aOff);
      __builtin_amdgcn_s_wait_tensorcnt(0);
    }
#else
    {  // fallback: manual coalesced staging with padded stride
      const unsigned int* src =
          (const unsigned int*)(neb + (size_t)(ob * RBLK) * DIM);
      unsigned int* dst = (unsigned int*)aLds;
#pragma unroll 4
      for (int i = 0; i < RBLK; ++i) {
        const int g = tid + 256 * i;
        dst[(g >> 8) * (STRIDE / 2) + (g & 255)] = src[g];
      }
    }
#endif
    __syncthreads();

    v8f acc[2][4] = {};
#pragma unroll
    for (int kk = 0; kk < DIM; kk += 32) {
      v16bf a[2], b[4];
#pragma unroll
      for (int rt = 0; rt < 2; ++rt)
        __builtin_memcpy(&a[rt], aP[rt] + kk, sizeof(v16bf));
#pragma unroll
      for (int ct = 0; ct < 4; ++ct)
        __builtin_memcpy(&b[ct], bP[ct] + kk, sizeof(v16bf));
#pragma unroll
      for (int rt = 0; rt < 2; ++rt)
#pragma unroll
        for (int ct = 0; ct < 4; ++ct)
          acc[rt][ct] = __builtin_amdgcn_wmma_f32_16x16x32_bf16(
              false, a[rt], false, b[ct], (short)0, acc[rt][ct], false, false);
    }

    // lean epilogue: clamp->scale, per-row max & sumexp over this wave's
    // 64 columns, then cross-wave merge over the 2 column groups
#pragma unroll
    for (int rt = 0; rt < 2; ++rt) {
#pragma unroll
      for (int r = 0; r < 8; ++r) {
        const int m = r + half * 8;                 // row in 16-tile
        const int lrow = rwave * 32 + rt * 16 + m;  // 0..127
        float v[4];
#pragma unroll
        for (int ct = 0; ct < 4; ++ct)
          v[ct] = (col[ct] < NCLS)
                      ? fminf(fmaxf(acc[rt][ct][r], -1.0f), 1.0f) * ARC_SCALE
                      : -INFINITY;
        float mx = fmaxf(fmaxf(v[0], v[1]), fmaxf(v[2], v[3]));
#pragma unroll
        for (int msk = 8; msk >= 1; msk >>= 1)
          mx = fmaxf(mx, __shfl_xor(mx, msk, 16));
        float e = 0.f;
#pragma unroll
        for (int ct = 0; ct < 4; ++ct)
          if (col[ct] < NCLS) e += __expf(v[ct] - mx);
#pragma unroll
        for (int msk = 8; msk >= 1; msk >>= 1) e += __shfl_xor(e, msk, 16);
        if (nl == 0) {
          wMax[cwave * RBLK + lrow] = mx;
          wSum[cwave * RBLK + lrow] = e;
        }
      }
    }
    __syncthreads();

    if (tid < RBLK) {
      float M = fmaxf(wMax[tid], wMax[RBLK + tid]);
      float S = 0.f;
#pragma unroll
      for (int cw = 0; cw < 2; ++cw) {
        const float s = wSum[cw * RBLK + tid];
        if (s > 0.f) S += s * __expf(wMax[cw * RBLK + tid] - M);
      }
      const size_t idx = (size_t)blockIdx.x * N_ROWS + ob * RBLK + tid;
      pmax[idx] = M;
      psum[idx] = S;
    }
    __syncthreads();
  }
}

// ---------------------------------------------------------------------------
// Kernel 3: merge per-chunk (max, sumexp) across 782 WGs, then swap the
// plain target term for the margin term inside the softmax denominator:
//   M2 = max(M, tm);  S2 = S*e^{M-M2} - e^{tp-M2} + e^{tm-M2}
//   loss_n = M2 + log(S2) - tm
// ---------------------------------------------------------------------------
__global__ void k_combine(const float* __restrict__ pmax,
                          const float* __restrict__ psum,
                          const float* __restrict__ tplain,
                          const float* __restrict__ tmarg,
                          float* __restrict__ rowloss) {
  const int wave = threadIdx.x >> 5, lane = threadIdx.x & 31;
  const int n = blockIdx.x * 8 + wave;
  float M = -INFINITY;
  for (int w = lane; w < NWG; w += 32)
    M = fmaxf(M, pmax[(size_t)w * N_ROWS + n]);
#pragma unroll
  for (int m = 16; m >= 1; m >>= 1) M = fmaxf(M, __shfl_xor(M, m, 32));
  float S = 0.f;
  for (int w = lane; w < NWG; w += 32) {
    const float mw = pmax[(size_t)w * N_ROWS + n];
    const float sw = psum[(size_t)w * N_ROWS + n];
    if (sw > 0.f) S += sw * __expf(mw - M);
  }
  S = waveReduceSum32(S);
  if (lane == 0) {
    const float tp = tplain[n];
    const float tm = tmarg[n];
    const float M2 = fmaxf(M, tm);
    float S2 = S * __expf(M - M2) - __expf(tp - M2) + __expf(tm - M2);
    S2 = fmaxf(S2, 1e-30f);
    rowloss[n] = (M2 + logf(S2)) - tm;
  }
}

// ---------------------------------------------------------------------------
// Kernel 4: mean over 1024 row losses -> scalar loss
// ---------------------------------------------------------------------------
__global__ void k_final(const float* __restrict__ rowloss,
                        float* __restrict__ out) {
  __shared__ float sm[256];
  float s = 0.f;
  for (int i = threadIdx.x; i < N_ROWS; i += 256) s += rowloss[i];
  sm[threadIdx.x] = s;
  __syncthreads();
  for (int st = 128; st >= 1; st >>= 1) {
    if (threadIdx.x < st) sm[threadIdx.x] += sm[threadIdx.x + st];
    __syncthreads();
  }
  if (threadIdx.x == 0) out[0] = sm[0] / (float)N_ROWS;
}

// ---------------------------------------------------------------------------
extern "C" void kernel_launch(void* const* d_in, const int* in_sizes, int n_in,
                              void* d_out, int out_size, void* d_ws,
                              size_t ws_size, hipStream_t stream) {
  (void)in_sizes; (void)n_in; (void)out_size; (void)ws_size;
  const float* images = (const float*)d_in[0];
  const int* labels = (const int*)d_in[1];
  const float* weight = (const float*)d_in[2];
  float* out = (float*)d_out;

  char* ws = (char*)d_ws;
  size_t off = 0;
  __bf16* neb = (__bf16*)(ws + off);   off += (size_t)N_ROWS * DIM * 2;
  float* tplain = (float*)(ws + off);  off += (size_t)N_ROWS * 4;
  float* tmarg = (float*)(ws + off);   off += (size_t)N_ROWS * 4;
  float* pmax = (float*)(ws + off);    off += (size_t)NWG * N_ROWS * 4;
  float* psum = (float*)(ws + off);    off += (size_t)NWG * N_ROWS * 4;
  float* rowloss = (float*)(ws + off); off += (size_t)N_ROWS * 4;

  // Allow ~268 KB dynamic LDS on CDNA5 (320 KB/WGP). Idempotent, capture-safe.
  hipFuncSetAttribute((const void*)k_arcface_main,
                      hipFuncAttributeMaxDynamicSharedMemorySize,
                      (int)SMEM_BYTES);

  k_norm_images<<<N_ROWS / 8, 256, 0, stream>>>(images, neb);
  k_target_logit<<<N_ROWS / 8, 256, 0, stream>>>(images, labels, weight,
                                                 tplain, tmarg);
  k_arcface_main<<<NWG, 256, SMEM_BYTES, stream>>>(weight, neb, pmax, psum);
  k_combine<<<N_ROWS / 8, 256, 0, stream>>>(pmax, psum, tplain, tmarg,
                                            rowloss);
  k_final<<<1, 256, 0, stream>>>(rowloss, out);
}
